// HybridPQC_24764781429239
// MI455X (gfx1250) — matrix operand order repaired and checked by reading, not compiled, not run on hardware
//
#include <hip/hip_runtime.h>
#include <hip/hip_bf16.h>
#include <math.h>

// ---------------------------------------------------------------------------
// Phase 1: h = tanh(X @ W^T) * scale  via v_wmma_f32_16x16x32_f16
//   X: (1024, 384) f32, W: (16, 384) f32 -> H: (1024, 16) f32
// One wave (32 lanes) computes one 16x16 output tile (16 rows x 16 qubits).
// ---------------------------------------------------------------------------

typedef __attribute__((ext_vector_type(16))) _Float16 v16h;
typedef __attribute__((ext_vector_type(8)))  float    v8f;

#define SBERT_DIM 384
#define NQ        16
#define NSTATE    (1 << NQ)      // 65536 amplitudes
#define NLAYERS   3

__global__ __launch_bounds__(32)
void pqc_angles_wmma(const float* __restrict__ X,
                     const float* __restrict__ W,
                     const float* __restrict__ scale,
                     float* __restrict__ H)
{
    const int tile = blockIdx.x;        // 64 tiles of 16 rows
    const int lane = threadIdx.x;       // 0..31 (wave32)
    const int hi   = lane >> 4;         // lane half: 0 or 1
    const int lo   = lane & 15;
    const int row  = tile * 16 + lo;    // A-matrix row for this lane

    v8f acc = {};

    for (int k0 = 0; k0 < SBERT_DIM; k0 += 32) {
        // A fragment: 16x32 f16, M=lo, ISA layout:
        //   VGPR j (j<4):  K = 2j   + 8*hi, +1
        //   VGPR j (j>=4): K = 16 + 2(j-4) + 8*hi, +1
        v16h a;
#pragma unroll
        for (int j = 0; j < 8; ++j) {
            const int kb = (j < 4) ? (2 * j + 8 * hi)
                                   : (16 + 2 * (j - 4) + 8 * hi);
            a[2 * j]     = (_Float16)X[row * SBERT_DIM + k0 + kb];
            a[2 * j + 1] = (_Float16)X[row * SBERT_DIM + k0 + kb + 1];
        }
        // B fragment: 32x16 f16, B[k][n] = W[n][k], N = lo.
        //   VGPR j (j<4):  K = 2j   + 8*hi, +1
        //   VGPR j (j>=4): K = 16 + 2(j-4) + 8*hi, +1
        v16h b;
#pragma unroll
        for (int j = 0; j < 8; ++j) {
            const int kb = (j < 4) ? (2 * j + 8 * hi)
                                   : (16 + 2 * (j - 4) + 8 * hi);
            b[2 * j]     = (_Float16)W[lo * SBERT_DIM + k0 + kb];
            b[2 * j + 1] = (_Float16)W[lo * SBERT_DIM + k0 + kb + 1];
        }
        acc = __builtin_amdgcn_wmma_f32_16x16x32_f16(
            /*neg_a=*/false, a, /*neg_b=*/false, b,
            /*c_mod=*/(short)0, acc, /*reuse_a=*/false, /*reuse_b=*/false);
    }

    const float s = scale[0];
    // D layout: VGPR j holds C[M = j + 8*hi][N = lo]
#pragma unroll
    for (int j = 0; j < 8; ++j) {
        const int m = j + 8 * hi;
        H[(tile * 16 + m) * NQ + lo] = tanhf(acc[j]) * s;
    }
}

// ---------------------------------------------------------------------------
// Phase 2: 16-qubit state-vector simulation, one workgroup per sample.
// State (256 KB fp32) lives entirely in LDS (320 KB/WGP on CDNA5).
// ---------------------------------------------------------------------------

__device__ __forceinline__ void ry_gate(float* state, int q, float c, float s,
                                        int tid, int nt)
{
    const int mlo = (1 << q) - 1;
    for (int p = tid; p < (NSTATE >> 1); p += nt) {
        const int i0 = ((p & ~mlo) << 1) | (p & mlo);
        const int i1 = i0 | (1 << q);
        const float a0 = state[i0];
        const float a1 = state[i1];
        state[i0] = c * a0 - s * a1;
        state[i1] = s * a0 + c * a1;
    }
    __syncthreads();
}

__device__ __forceinline__ void cnot_gate(float* state, int ctrl, int tgt,
                                          int tid, int nt)
{
    const int lo  = (ctrl < tgt) ? ctrl : tgt;
    const int hq  = (ctrl < tgt) ? tgt : ctrl;
    const int mid = hq - lo - 1;
    for (int p = tid; p < (NSTATE >> 2); p += nt) {
        int x = p;
        int i = x & ((1 << lo) - 1);
        x >>= lo;
        i |= (x & ((1 << mid) - 1)) << (lo + 1);
        x >>= mid;
        i |= x << (hq + 1);
        i |= (1 << ctrl);              // ctrl bit = 1, tgt bit = 0
        const int j = i | (1 << tgt);
        const float t = state[i];
        state[i] = state[j];
        state[j] = t;
    }
    __syncthreads();
}

__global__ __launch_bounds__(1024)
void pqc_sim(const float* __restrict__ H,        // (B, 16) angles
             const float* __restrict__ Wt,       // (3, 16) layer weights
             float* __restrict__ out)            // (B, 16) <Z_q>
{
    extern __shared__ float smem[];
    float* state = smem;                 // 65536 floats
    float* ez    = smem + NSTATE;        // 16 accumulators

    const int b   = blockIdx.x;
    const int tid = threadIdx.x;
    const int nt  = blockDim.x;          // 1024

    for (int i = tid; i < NSTATE; i += nt) state[i] = (i == 0) ? 1.0f : 0.0f;
    if (tid < NQ) ez[tid] = 0.0f;
    __syncthreads();

    // Input encoding: RY(h[b][q]) on each qubit
    for (int q = 0; q < NQ; ++q) {
        float s, c;
        __sincosf(0.5f * H[b * NQ + q], &s, &c);
        ry_gate(state, q, c, s, tid, nt);
    }

    // Variational layers: RY(weights[l][q]) then linear CNOT chain
    for (int l = 0; l < NLAYERS; ++l) {
        for (int q = 0; q < NQ; ++q) {
            float s, c;
            __sincosf(0.5f * Wt[l * NQ + q], &s, &c);
            ry_gate(state, q, c, s, tid, nt);
        }
        for (int q = 0; q < NQ - 1; ++q)
            cnot_gate(state, q, q + 1, tid, nt);
    }

    // <Z_q> = sum_i (+-) state[i]^2, sign = bit q of i
    float acc[NQ];
#pragma unroll
    for (int q = 0; q < NQ; ++q) acc[q] = 0.0f;

    for (int i = tid; i < NSTATE; i += nt) {
        const float a = state[i];
        const float p = a * a;
#pragma unroll
        for (int q = 0; q < NQ; ++q)
            acc[q] += ((i >> q) & 1) ? -p : p;
    }

    // wave32 reduction, then one shared atomic per wave per qubit
#pragma unroll
    for (int q = 0; q < NQ; ++q) {
        float v = acc[q];
        for (int off = 16; off > 0; off >>= 1)
            v += __shfl_down(v, off, 32);
        if ((tid & 31) == 0) atomicAdd(&ez[q], v);
    }
    __syncthreads();

    if (tid < NQ) out[b * NQ + tid] = ez[tid];
}

// ---------------------------------------------------------------------------
// Launch
// ---------------------------------------------------------------------------

extern "C" void kernel_launch(void* const* d_in, const int* in_sizes, int n_in,
                              void* d_out, int out_size, void* d_ws, size_t ws_size,
                              hipStream_t stream)
{
    const float* X     = (const float*)d_in[0];   // (1024, 384)
    const float* W     = (const float*)d_in[1];   // (16, 384)
    const float* scale = (const float*)d_in[2];   // scalar
    const float* wts   = (const float*)d_in[3];   // (3, 16)
    float* out = (float*)d_out;                   // (1024, 16)

    const int BATCH = in_sizes[0] / SBERT_DIM;    // 1024
    float* H = (float*)d_ws;                      // (BATCH, 16) angles scratch

    // Phase 1: WMMA projection, one wave per 16-row tile
    pqc_angles_wmma<<<BATCH / 16, 32, 0, stream>>>(X, W, scale, H);

    // Phase 2: LDS-resident state-vector sim, one workgroup per sample
    const size_t smem_bytes = (size_t)(NSTATE + NQ) * sizeof(float); // 262,208 B
    static_assert((NSTATE + NQ) * sizeof(float) <= 320 * 1024, "fits WGP LDS");
    (void)hipFuncSetAttribute((const void*)pqc_sim,
                              hipFuncAttributeMaxDynamicSharedMemorySize,
                              (int)smem_bytes);
    pqc_sim<<<BATCH, 1024, smem_bytes, stream>>>(H, wts, out);
}
